// MultiHeadedAttention_56831007261350
// MI455X (gfx1250) — matrix operand order repaired and checked by reading, not compiled, not run on hardware
//
#include <hip/hip_runtime.h>
#include <hip/hip_bf16.h>
#include <stdint.h>

// ---------------------------------------------------------------------------
// MHA forward for MI455X (gfx1250, wave32, WMMA 16x16x32 f16 -> f32 acc)
// B=2, S=2048, D_MODEL=1024, H=16, DK=64
//
// Pipeline: (0) f32->f16 convert of activations+weights (bandwidth-trivial),
// (1) QKV projections via double-buffered async-LDS WMMA GEMM,
// (2) flash attention (WMMA scores + online softmax + WMMA PV),
// (3) output projection GEMM -> f32.
// ---------------------------------------------------------------------------

typedef _Float16 f16;
typedef _Float16 v16h __attribute__((ext_vector_type(16)));
typedef _Float16 v8h  __attribute__((ext_vector_type(8)));
typedef float    v8f  __attribute__((ext_vector_type(8)));
typedef int      v4i  __attribute__((vector_size(16)));   // matches builtin proto

#define D_MODEL 1024
#define NHEAD   16
#define DK      64
#define SEQ     2048
#define BATCH   2

// GEMM tiling: block tile 128(M) x 64(N), K-step 32, 256 threads = 8 waves.
#define BM 128
#define BN 64
#define BK 32
#define LDA 40   // 32 + 8 pad halfs; row stride 80B (16B multiple)
#define LDB 40

// ---------------------------------------------------------------------------
// Async global->LDS copy (16B per lane), ASYNCcnt-tracked. Fallback: sync v8h.
// Builtin proto (from clang diagnostic): param0 = v4i __device__* (global),
// param1 = v4i __shared__* (LDS), then imm offset, imm cpol.
// ---------------------------------------------------------------------------
#if defined(__gfx1250__) && __has_builtin(__builtin_amdgcn_global_load_async_to_lds_b128)
#define HAS_ASYNC_LDS 1
#else
#define HAS_ASYNC_LDS 0
#endif

#if HAS_ASYNC_LDS
typedef __attribute__((address_space(1))) v4i* gas_v4i_ptr;
typedef __attribute__((address_space(3))) v4i* las_v4i_ptr;
#endif

__device__ __forceinline__ void async_cp16(void* lds_dst, const void* gsrc) {
#if HAS_ASYNC_LDS
    // Global flat address == AS1 address; LDS offset = low 32 bits of flat
    // shared address.
    __builtin_amdgcn_global_load_async_to_lds_b128(
        (gas_v4i_ptr)(uintptr_t)gsrc,
        (las_v4i_ptr)(uint32_t)(uintptr_t)lds_dst,
        0, 0);
#else
    *(v8h*)lds_dst = *(const v8h*)gsrc;
#endif
}

template <int N>
__device__ __forceinline__ void wait_async() {
#if HAS_ASYNC_LDS
#if __has_builtin(__builtin_amdgcn_s_wait_asynccnt)
    __builtin_amdgcn_s_wait_asynccnt(N);
#else
    asm volatile("s_wait_asynccnt %0" :: "i"(N) : "memory");
#endif
#endif
}

// ---------------------------------------------------------------------------
// Kernel 0: elementwise f32 -> f16 (8 elements per thread, vectorized).
// ---------------------------------------------------------------------------
__global__ __launch_bounds__(256)
void cvt_f32_to_f16(const float* __restrict__ src, f16* __restrict__ dst, int n8) {
    int i = blockIdx.x * 256 + threadIdx.x;
    if (i < n8) {
        const float4* s = (const float4*)src + (size_t)i * 2;
        float4 a = s[0], b = s[1];
        v8h o;
        o[0] = (f16)a.x; o[1] = (f16)a.y; o[2] = (f16)a.z; o[3] = (f16)a.w;
        o[4] = (f16)b.x; o[5] = (f16)b.y; o[6] = (f16)b.z; o[7] = (f16)b.w;
        *((v8h*)dst + i) = o;
    }
}

// ---------------------------------------------------------------------------
// Kernel 1/3: WMMA GEMM  Y = X @ W^T (+ bias), f16 inputs staged to LDS with
// double-buffered async copies.
//   SCATTER=true : Y -> f16 [B,H,S,DK] (QKV), bias added pre-cast.
//   SCATTER=false: Y -> f32 [B*S, D_MODEL] with bias (output projection).
// ---------------------------------------------------------------------------
template <bool SCATTER>
__global__ __launch_bounds__(256)
void gemm16_kernel(const f16* __restrict__ X, const f16* __restrict__ W,
                   const float* __restrict__ bias,
                   f16* __restrict__ Yh, float* __restrict__ Yf)
{
    __shared__ __attribute__((aligned(16))) f16 sX[2][BM][LDA];
    __shared__ __attribute__((aligned(16))) f16 sW[2][BN][LDB];

    const int tid  = threadIdx.x;
    const int wave = tid >> 5;
    const int lane = tid & 31;
    const int hl   = lane >> 4;
    const int ln16 = lane & 15;

    const int m0 = blockIdx.x * BM;
    const int n0 = blockIdx.y * BN;

    // Per-iteration staging: X 128x32 halfs = 512 x 16B (2/thread),
    //                        W  64x32 halfs = 256 x 16B (1/thread). 3 asyncs/thread.
    auto stage = [&](int k0, int buf) {
        for (int i = 0; i < 2; ++i) {
            int j   = i * 256 + tid;     // 0..511
            int row = j >> 2;            // 4 chunks per 32-half row
            int c   = j & 3;
            async_cp16(&sX[buf][row][c * 8],
                       X + (size_t)(m0 + row) * D_MODEL + k0 + c * 8);
        }
        {
            int row = tid >> 2;          // 0..63
            int c   = tid & 3;
            async_cp16(&sW[buf][row][c * 8],
                       W + (size_t)(n0 + row) * D_MODEL + k0 + c * 8);
        }
    };

    stage(0, 0);

    v8f acc[4] = {};

    for (int k0 = 0, it = 0; k0 < D_MODEL; k0 += BK, ++it) {
        const int buf = it & 1;
        if (k0 + BK < D_MODEL) {
            stage(k0 + BK, buf ^ 1);   // prefetch next tile into other buffer
            wait_async<3>();           // this iter's 3 older copies complete
        } else {
            wait_async<0>();
        }
        __syncthreads();

        // A fragment (16x32): lane<16 holds K={0..7,16..23}, lane>=16 K={8..15,24..31}.
        union { v16h v; v8h hh[2]; } a;
        const f16* arow = &sX[buf][wave * 16 + ln16][0];
        a.hh[0] = *(const v8h*)(arow + 8 * hl);
        a.hh[1] = *(const v8h*)(arow + 16 + 8 * hl);

        for (int nt = 0; nt < 4; ++nt) {
            // B fragment (32x16): lane<16 holds K={0..7,16..23}, lane>=16 K={8..15,24..31}, col = ln16.
            union { v16h v; v8h hh[2]; } b;
            const f16* brow = &sW[buf][nt * 16 + ln16][0];
            b.hh[0] = *(const v8h*)(brow + 8 * hl);
            b.hh[1] = *(const v8h*)(brow + 16 + 8 * hl);
            acc[nt] = __builtin_amdgcn_wmma_f32_16x16x32_f16(
                false, a.v, false, b.v, (short)0, acc[nt], false, false);
        }
        __syncthreads();   // readers done before buffer is overwritten
    }

    // Epilogue. C layout: row = r + 8*hl, col = ln16.
    for (int nt = 0; nt < 4; ++nt) {
        int col = n0 + nt * 16 + ln16;
        float bval = bias[col];
        if (SCATTER) {
            int hh = col >> 6, dd = col & 63;
            for (int r = 0; r < 8; ++r) {
                int row = m0 + wave * 16 + r + 8 * hl;
                int bb = row >> 11, ss = row & (SEQ - 1);
                Yh[(((size_t)(bb * NHEAD + hh) * SEQ + ss) << 6) + dd] =
                    (f16)(acc[nt][r] + bval);
            }
        } else {
            for (int r = 0; r < 8; ++r) {
                int row = m0 + wave * 16 + r + 8 * hl;
                Yf[(size_t)row * D_MODEL + col] = acc[nt][r] + bval;
            }
        }
    }
}

// ---------------------------------------------------------------------------
// Kernel 2: flash attention. One block per (b,h, 64-row q tile); 4 waves,
// each wave owns 16 q rows. K/V consumed in 32-column blocks.
// ---------------------------------------------------------------------------
#define QT 64
#define KT 32

__global__ __launch_bounds__(128)
void attn_kernel(const f16* __restrict__ Qh, const f16* __restrict__ Kh,
                 const f16* __restrict__ Vh, const int* __restrict__ mask,
                 f16* __restrict__ att)
{
    const int qblk = blockIdx.x;           // 0..S/QT-1
    const int bh   = blockIdx.y;           // 0..B*H-1
    const int b    = bh >> 4, h = bh & 15;

    __shared__ __attribute__((aligned(16))) f16 sQ[QT][DK + 8];    // 64x72
    __shared__ __attribute__((aligned(16))) f16 sK[KT][DK + 8];    // 32x72 ([k'][d])
    __shared__ __attribute__((aligned(16))) f16 sVt[DK][KT + 8];   // 64x40 ([d][k'])
    __shared__ __attribute__((aligned(16))) f16 sP[4][16][KT + 8]; // per-wave P staging

    const int tid  = threadIdx.x;
    const int wave = tid >> 5, lane = tid & 31;
    const int hl   = lane >> 4, ln16 = lane & 15;

    const size_t base = (size_t)(b * NHEAD + h) * SEQ * DK;
    const f16* Qp = Qh + base + (size_t)qblk * QT * DK;
    const f16* Kp = Kh + base;
    const f16* Vp = Vh + base;
    const int* mp = mask + b * SEQ;

    // Async-stage Q tile: 64x64 halfs = 512 x 16B, 4 per thread.
    for (int i = 0; i < 4; ++i) {
        int j = i * 128 + tid;
        int row = j >> 3, c8 = j & 7;
        async_cp16(&sQ[row][c8 * 8], Qp + row * DK + c8 * 8);
    }
    wait_async<0>();
    __syncthreads();

    float mrow[8], lrow[8];
    v8f O[4] = {};
    for (int r = 0; r < 8; ++r) { mrow[r] = -1e30f; lrow[r] = 0.0f; }
    const float scale = 0.125f;  // 1/sqrt(DK)

    for (int kb = 0; kb < SEQ; kb += KT) {
        // Warm GL2 for the next block (global_prefetch_b8).
        if (kb + KT < SEQ) {
            __builtin_prefetch(Kp + (size_t)(kb + KT) * DK + (tid & 15) * 128, 0, 1);
            __builtin_prefetch(Vp + (size_t)(kb + KT) * DK + (tid & 15) * 128, 0, 1);
        }
        // Async-stage K block (32x64): 256 x 16B, 2 per thread.
        for (int i = 0; i < 2; ++i) {
            int j = i * 128 + tid;
            int row = j >> 3, c8 = j & 7;
            async_cp16(&sK[row][c8 * 8], Kp + (size_t)(kb + row) * DK + c8 * 8);
        }
        // Stage V block transposed (VALU path): sVt[d][k'] = V[kb+k'][d].
        {
            int row = tid >> 2;            // 0..31  (k')
            int d0  = (tid & 3) * 16;      // 0,16,32,48
            for (int d = 0; d < 16; ++d)
                sVt[d0 + d][row] = Vp[(size_t)(kb + row) * DK + d0 + d];
        }
        wait_async<0>();
        __syncthreads();

        // Scores: S = Q(16xDK) . K^T  -> two 16x16 f32 tiles (columns kb..kb+31).
        v8f sc[2] = {};
        for (int nt = 0; nt < 2; ++nt) {
            for (int ks = 0; ks < 2; ++ks) {
                union { v16h v; v8h hh[2]; } af, bf;
                const f16* ar = &sQ[wave * 16 + ln16][ks * 32];
                af.hh[0] = *(const v8h*)(ar + 8 * hl);
                af.hh[1] = *(const v8h*)(ar + 16 + 8 * hl);
                const f16* br = &sK[nt * 16 + ln16][ks * 32];
                bf.hh[0] = *(const v8h*)(br + 8 * hl);
                bf.hh[1] = *(const v8h*)(br + 16 + 8 * hl);
                sc[nt] = __builtin_amdgcn_wmma_f32_16x16x32_f16(
                    false, af.v, false, bf.v, (short)0, sc[nt], false, false);
            }
        }

        // Scale + mask (mask[b][0][k]; shared by all q rows).
        for (int nt = 0; nt < 2; ++nt) {
            int kcol = kb + nt * 16 + ln16;
            float mb = (mp[kcol] == 0) ? -1.0e9f : 0.0f;
            for (int r = 0; r < 8; ++r) sc[nt][r] = sc[nt][r] * scale + mb;
        }

        // Online softmax. C layout puts a row's 16 cols across one 16-lane half,
        // so xor masks 1,2,4,8 reduce exactly over one row.
        float corr[8];
        for (int r = 0; r < 8; ++r) {
            float v = fmaxf(sc[0][r], sc[1][r]);
            v = fmaxf(v, __shfl_xor(v, 1));
            v = fmaxf(v, __shfl_xor(v, 2));
            v = fmaxf(v, __shfl_xor(v, 4));
            v = fmaxf(v, __shfl_xor(v, 8));
            float newm = fmaxf(mrow[r], v);
            corr[r] = expf(mrow[r] - newm);
            mrow[r] = newm;
        }
        // p = exp(s - m); stage to per-wave LDS region in A-source layout.
        for (int nt = 0; nt < 2; ++nt) {
            for (int r = 0; r < 8; ++r) {
                float p = expf(sc[nt][r] - mrow[r]);
                sc[nt][r] = p;
                sP[wave][r + 8 * hl][nt * 16 + ln16] = (f16)p;
            }
        }
        for (int r = 0; r < 8; ++r) {
            float rs = sc[0][r] + sc[1][r];
            rs += __shfl_xor(rs, 1);
            rs += __shfl_xor(rs, 2);
            rs += __shfl_xor(rs, 4);
            rs += __shfl_xor(rs, 8);
            lrow[r] = lrow[r] * corr[r] + rs;
        }
        for (int t = 0; t < 4; ++t)
            for (int r = 0; r < 8; ++r) O[t][r] *= corr[r];

        // O += P(16x32) @ V(32x64). Wave-private sP; DS ops in-order per wave.
        union { v16h v; v8h hh[2]; } pf;
        const f16* pr = &sP[wave][ln16][0];
        pf.hh[0] = *(const v8h*)(pr + 8 * hl);
        pf.hh[1] = *(const v8h*)(pr + 16 + 8 * hl);
        for (int t = 0; t < 4; ++t) {
            union { v16h v; v8h hh[2]; } vf;
            const f16* vr = &sVt[t * 16 + ln16][0];
            vf.hh[0] = *(const v8h*)(vr + 8 * hl);
            vf.hh[1] = *(const v8h*)(vr + 16 + 8 * hl);
            O[t] = __builtin_amdgcn_wmma_f32_16x16x32_f16(
                false, pf.v, false, vf.v, (short)0, O[t], false, false);
        }
        __syncthreads();
    }

    // Normalize and write attn output as f16 in [B, S, H*DK].
    for (int r = 0; r < 8; ++r) mrow[r] = 1.0f / lrow[r];  // reuse as inverse
    for (int t = 0; t < 4; ++t) {
        int d = t * 16 + ln16;
        for (int r = 0; r < 8; ++r) {
            int srow = qblk * QT + wave * 16 + r + 8 * hl;
            att[(size_t)(b * SEQ + srow) * D_MODEL + h * DK + d] =
                (f16)(O[t][r] * mrow[r]);
        }
    }
}

// ---------------------------------------------------------------------------
// Host launch
// ---------------------------------------------------------------------------
extern "C" void kernel_launch(void* const* d_in, const int* in_sizes, int n_in,
                              void* d_out, int out_size, void* d_ws, size_t ws_size,
                              hipStream_t stream) {
    (void)in_sizes; (void)n_in; (void)out_size; (void)ws_size;

    const float* query = (const float*)d_in[0];
    const float* key   = (const float*)d_in[1];
    const float* value = (const float*)d_in[2];
    const int*   mask  = (const int*)  d_in[3];
    const float* Wq = (const float*)d_in[4];  const float* bq = (const float*)d_in[5];
    const float* Wk = (const float*)d_in[6];  const float* bk = (const float*)d_in[7];
    const float* Wv = (const float*)d_in[8];  const float* bv = (const float*)d_in[9];
    const float* Wo = (const float*)d_in[10]; const float* bo = (const float*)d_in[11];

    const size_t n_x = (size_t)BATCH * SEQ * D_MODEL;   // 4.19M elems
    const size_t n_w = (size_t)D_MODEL * D_MODEL;       // 1.05M elems

    f16* xq = (f16*)d_ws;
    f16* xk = xq + n_x;
    f16* xv = xk + n_x;
    f16* wq = xv + n_x;
    f16* wk = wq + n_w;
    f16* wv = wk + n_w;
    f16* wo = wv + n_w;
    f16* qh = wo + n_w;
    f16* kh = qh + n_x;
    f16* vh = kh + n_x;
    f16* att = vh + n_x;

    // (0) converts: activations + weights to f16 (bandwidth-trivial).
    {
        int nx8 = (int)(n_x / 8), nw8 = (int)(n_w / 8);
        cvt_f32_to_f16<<<nx8 / 256, 256, 0, stream>>>(query, xq, nx8);
        cvt_f32_to_f16<<<nx8 / 256, 256, 0, stream>>>(key,   xk, nx8);
        cvt_f32_to_f16<<<nx8 / 256, 256, 0, stream>>>(value, xv, nx8);
        cvt_f32_to_f16<<<nw8 / 256, 256, 0, stream>>>(Wq, wq, nw8);
        cvt_f32_to_f16<<<nw8 / 256, 256, 0, stream>>>(Wk, wk, nw8);
        cvt_f32_to_f16<<<nw8 / 256, 256, 0, stream>>>(Wv, wv, nw8);
        cvt_f32_to_f16<<<nw8 / 256, 256, 0, stream>>>(Wo, wo, nw8);
    }

    // (1) QKV projections -> [B,H,S,DK] f16.
    dim3 g1(BATCH * SEQ / BM, D_MODEL / BN);   // 32 x 16
    gemm16_kernel<true><<<g1, 256, 0, stream>>>(xq, wq, bq, qh, nullptr);
    gemm16_kernel<true><<<g1, 256, 0, stream>>>(xk, wk, bk, kh, nullptr);
    gemm16_kernel<true><<<g1, 256, 0, stream>>>(xv, wv, bv, vh, nullptr);

    // (2) attention.
    dim3 g2(SEQ / QT, BATCH * NHEAD);          // 32 x 32
    attn_kernel<<<g2, 128, 0, stream>>>(qh, kh, vh, mask, att);

    // (3) output projection -> f32 d_out.
    gemm16_kernel<false><<<g1, 256, 0, stream>>>(att, wo, bo, nullptr, (float*)d_out);
}
